// MultiDConvHeadTransposedAttention_70411693851262
// MI455X (gfx1250) — matrix operand order repaired and checked by reading, not compiled, not run on hardware
//
#include <hip/hip_runtime.h>
#include <hip/hip_bf16.h>
#include <math.h>
#include <stdint.h>

typedef __attribute__((ext_vector_type(16))) _Float16 v16h;
typedef __attribute__((ext_vector_type(8)))  _Float16 v8h;
typedef __attribute__((ext_vector_type(8)))  float    v8f;

#define B_    8
#define C_    384
#define HW_   16384        // 128*128
#define P_    131072       // B_*HW_  (= 2^17)
#define O1_   1152         // 3*C
#define EPS_  1e-5f

// workspace byte offsets (256-aligned)
#define XN_OFF    0ull            // f16 [P][384]   (reused as attn[P][384] later)
#define WP_OFF    100663296ull    // f16 [1152][384]
#define WO_OFF    101548032ull    // f16 [384][384]
#define SF_OFF    101842944ull    // f32 [64][48][48]
#define SHT_OFF   102432768ull    // f16 [64][64][64]  S transposed+padded
#define QKV1_OFF  102957056ull    // f16 [P][1152]
#define QKV2_OFF  404946944ull    // f16 [P][1152]

static __device__ __forceinline__ v16h cat16(v8h a, v8h b) {
  return __builtin_shufflevector(a, b, 0,1,2,3,4,5,6,7,8,9,10,11,12,13,14,15);
}

// low 32 bits of a flat LDS pointer = wave-relative LDS byte address
static __device__ __forceinline__ unsigned ldsAddrOf(const void* p) {
  return (unsigned)(uintptr_t)p;
}
// async 16B copy global -> LDS (tracked with ASYNCcnt, in-order completion)
static __device__ __forceinline__ void asyncCopy16(unsigned lds, const void* g) {
  asm volatile("global_load_async_to_lds_b128 %0, %1, off"
               :: "v"(lds), "v"(g) : "memory");
}
#define WAIT_ASYNC(n) asm volatile("s_wait_asynccnt " #n ::: "memory")

// ---------------- LayerNorm over channel dim, write xn[p][c] (f16) ----------
__global__ void k_ln(const float* __restrict__ x, const float* __restrict__ lnw,
                     _Float16* __restrict__ xn) {
  __shared__ float sw[C_];
  int tid = threadIdx.x;
  for (int i = tid; i < C_; i += 256) sw[i] = lnw[i];
  __syncthreads();
  int p  = blockIdx.x * 256 + tid;
  int b  = p >> 14;
  int hw = p & (HW_ - 1);
  const float* xp = x + (size_t)b * C_ * HW_ + hw;
  float s = 0.f, s2 = 0.f;
  for (int c = 0; c < C_; ++c) { float v = xp[(size_t)c * HW_]; s += v; s2 += v * v; }
  float mean = s * (1.f / C_);
  float var  = s2 * (1.f / C_) - mean * mean;
  float rstd = rsqrtf(var + EPS_);
  _Float16* op = xn + (size_t)p * C_;
  for (int c0 = 0; c0 < C_; c0 += 8) {
    v8h vv;
    #pragma unroll
    for (int t = 0; t < 8; ++t)
      vv[t] = (_Float16)((xp[(size_t)(c0 + t) * HW_] - mean) * rstd * sw[c0 + t]);
    *(v8h*)(op + c0) = vv;
  }
}

// ---------------- weights f32 -> f16 ----------------------------------------
__global__ void k_cvt(const float* __restrict__ wp, const float* __restrict__ wo,
                      _Float16* __restrict__ wph, _Float16* __restrict__ woh) {
  int i = blockIdx.x * 256 + threadIdx.x;
  if (i < O1_ * C_) {
    wph[i] = (_Float16)wp[i];
  } else {
    int j = i - O1_ * C_;
    if (j < C_ * C_) woh[j] = (_Float16)wo[j];
  }
}

// ---- QKV GEMM: Out[p][o] = sum_k X[p][k]*W[o][k]; async double-buffered ----
__global__ void __launch_bounds__(256) k_gemm_qkv(const _Float16* __restrict__ Wh,
                                                  const _Float16* __restrict__ Xh,
                                                  _Float16* __restrict__ Out) {
  __shared__ __align__(16) unsigned short lsA[2][64 * 40];  // X tile [p][k]
  __shared__ __align__(16) unsigned short lsB[2][64 * 40];  // W tile [o][k]
  int tid  = threadIdx.x;
  int lane = tid & 31, wv = tid >> 5;
  int hi   = lane >> 4;
  int wm   = wv & 3, wn = wv >> 2;
  int p0   = blockIdx.x * 64;
  int o0   = blockIdx.y * 64;
  // staging: one 16B chunk of A and one of B per thread per stage
  int sr = tid >> 2, sc = tid & 3;
  const _Float16* gA = Xh + (size_t)(p0 + sr) * 384 + sc * 8;
  const _Float16* gB = Wh + (size_t)(o0 + sr) * 384 + sc * 8;
  unsigned la[2] = { ldsAddrOf(&lsA[0][sr * 40 + sc * 8]),
                     ldsAddrOf(&lsA[1][sr * 40 + sc * 8]) };
  unsigned lb[2] = { ldsAddrOf(&lsB[0][sr * 40 + sc * 8]),
                     ldsAddrOf(&lsB[1][sr * 40 + sc * 8]) };
  asyncCopy16(la[0], gA);
  asyncCopy16(lb[0], gB);
  v8f acc0 = {}; v8f acc1 = {};
  for (int i = 0; i < 12; ++i) {
    if (i < 11) {
      int nb = (i + 1) & 1;
      asyncCopy16(la[nb], gA + (i + 1) * 32);
      asyncCopy16(lb[nb], gB + (i + 1) * 32);
      WAIT_ASYNC(0x2);
    } else {
      WAIT_ASYNC(0x0);
    }
    __syncthreads();
    int cb = i & 1;
    const _Float16* Af = (const _Float16*)lsA[cb];
    const _Float16* Bf = (const _Float16*)lsB[cb];
    int ar = (wm * 16 + (lane & 15)) * 40;
    v16h a = cat16(*(const v8h*)(Af + ar + hi * 16),
                   *(const v8h*)(Af + ar + hi * 16 + 8));
    int br0 = (wn * 32 + (lane & 15)) * 40 + hi * 16;
    v16h b0 = cat16(*(const v8h*)(Bf + br0), *(const v8h*)(Bf + br0 + 8));
    int br1 = (wn * 32 + 16 + (lane & 15)) * 40 + hi * 16;
    v16h b1 = cat16(*(const v8h*)(Bf + br1), *(const v8h*)(Bf + br1 + 8));
    acc0 = __builtin_amdgcn_wmma_f32_16x16x32_f16(false, a, false, b0, (short)0, acc0, false, false);
    acc1 = __builtin_amdgcn_wmma_f32_16x16x32_f16(false, a, false, b1, (short)0, acc1, false, false);
    __syncthreads();
  }
  #pragma unroll
  for (int r = 0; r < 8; ++r) {
    size_t p = (size_t)(p0 + wm * 16 + hi * 8 + r);
    int   o  = o0 + wn * 32 + (lane & 15);
    Out[p * O1_ + o]      = (_Float16)acc0[r];
    Out[p * O1_ + o + 16] = (_Float16)acc1[r];
  }
}

// ---------------- depthwise 3x3, zero pad, layout [p][c] ---------------------
__global__ void k_dwconv(const _Float16* __restrict__ src, const float* __restrict__ wd,
                         _Float16* __restrict__ dst) {
  size_t e = (size_t)blockIdx.x * 256 + threadIdx.x;   // e = p*1152 + o
  int p = (int)(e / O1_);
  int o = (int)(e - (size_t)p * O1_);
  int hw = p & (HW_ - 1);
  int hh = hw >> 7, ww = hw & 127;
  const _Float16* sp = src + (size_t)(p - hw) * O1_ + o;
  const float* w = wd + o * 9;
  float acc = 0.f;
  #pragma unroll
  for (int dy = 0; dy < 3; ++dy) {
    int y = hh + dy - 1;
    if ((unsigned)y < 128u) {
      #pragma unroll
      for (int dx = 0; dx < 3; ++dx) {
        int xx = ww + dx - 1;
        if ((unsigned)xx < 128u)
          acc += (float)sp[(size_t)(y * 128 + xx) * O1_] * w[dy * 3 + dx];
      }
    }
  }
  dst[e] = (_Float16)acc;
}

// ---------------- S[i][j] = sum_n q[n,i]*k[n,j], per (b,head) ----------------
__global__ void __launch_bounds__(288) k_score(const _Float16* __restrict__ qkv,
                                               float* __restrict__ Sf) {
  __shared__ __align__(16) unsigned short lsA[48 * 40];  // q^T tile [i][n]
  __shared__ __align__(16) unsigned short lsB[48 * 40];  // k^T tile [j][n]
  const unsigned int* Qu = (const unsigned int*)qkv;
  int tid  = threadIdx.x;
  int lane = tid & 31, wv = tid >> 5;      // 9 waves
  int hi   = lane >> 4;
  int mi   = wv / 3, nj = wv % 3;
  int bh   = blockIdx.x;
  int b = bh >> 3, h = bh & 7;
  v8f acc = {};
  for (int kk = 0; kk < HW_; kk += 32) {
    // transpose-stage: 32 n-rows; q part (24 u32) and k part (24 u32) per row
    for (int idx = tid; idx < 1536; idx += 288) {
      int t = idx & 767;
      int n = t / 24, c = t % 24;
      size_t row = (size_t)(b * HW_ + kk + n) * 576;   // u32 per row = 1152/2
      if (idx < 768) {
        unsigned v = Qu[row + h * 24 + c];
        lsA[(2 * c) * 40 + n]     = (unsigned short)(v & 0xffff);
        lsA[(2 * c + 1) * 40 + n] = (unsigned short)(v >> 16);
      } else {
        unsigned v = Qu[row + 192 + h * 24 + c];       // k channels at +384 halves
        lsB[(2 * c) * 40 + n]     = (unsigned short)(v & 0xffff);
        lsB[(2 * c + 1) * 40 + n] = (unsigned short)(v >> 16);
      }
    }
    __syncthreads();
    const _Float16* Af = (const _Float16*)lsA;
    const _Float16* Bf = (const _Float16*)lsB;
    int ar = (mi * 16 + (lane & 15)) * 40;
    v16h a = cat16(*(const v8h*)(Af + ar + hi * 16),
                   *(const v8h*)(Af + ar + hi * 16 + 8));
    int br = (nj * 16 + (lane & 15)) * 40 + hi * 16;
    v16h bf = cat16(*(const v8h*)(Bf + br), *(const v8h*)(Bf + br + 8));
    acc = __builtin_amdgcn_wmma_f32_16x16x32_f16(false, a, false, bf, (short)0, acc, false, false);
    __syncthreads();
  }
  float* out = Sf + (size_t)bh * 2304;
  #pragma unroll
  for (int r = 0; r < 8; ++r) {
    int i = mi * 16 + hi * 8 + r;
    int j = nj * 16 + (lane & 15);
    out[i * 48 + j] = acc[r];
  }
}

// ---------------- softmax(S/alpha) rows, write S^T padded 64x64 f16 ----------
__global__ void k_softmax(const float* __restrict__ Sf, const float* __restrict__ alpha,
                          _Float16* __restrict__ ShT) {
  int t  = blockIdx.x * 256 + threadIdx.x;  // 4096 = 64 bh * 64 rows
  int bh = t >> 6, i = t & 63;
  _Float16* outp = ShT + (size_t)bh * 4096;
  if (i >= 48) {
    for (int j = 0; j < 64; ++j) outp[j * 64 + i] = (_Float16)0.f;
    return;
  }
  float inv = 1.f / alpha[0];
  const float* row = Sf + (size_t)bh * 2304 + i * 48;
  float mz = -3.4e38f;
  for (int j = 0; j < 48; ++j) mz = fmaxf(mz, row[j] * inv);
  float sum = 0.f;
  for (int j = 0; j < 48; ++j) sum += __expf(row[j] * inv - mz);
  float rs = 1.f / sum;
  for (int j = 0; j < 48; ++j) outp[j * 64 + i] = (_Float16)(__expf(row[j] * inv - mz) * rs);
  for (int j = 48; j < 64; ++j) outp[j * 64 + i] = (_Float16)0.f;
}

// ------- attn[n][j] = sum_i v[n,i]*S[i,j]; both tiles async-staged ----------
__global__ void __launch_bounds__(384) k_attnv(const _Float16* __restrict__ qkv,
                                               const _Float16* __restrict__ ShT,
                                               _Float16* __restrict__ attn) {
  __shared__ __align__(16) unsigned short lsA[64 * 72];  // v tile [n][i]
  __shared__ __align__(16) unsigned short lsB[64 * 72];  // S^T tile [j][i]
  int tid  = threadIdx.x;                  // 384 = 12 waves
  int lane = tid & 31, wv = tid >> 5;
  int hi   = lane >> 4;
  int wm   = wv / 3, wj = wv % 3;
  int bh   = blockIdx.y;
  int b = bh >> 3, h = bh & 7;
  int n0   = blockIdx.x * 64;
  // A: 64 n-rows x 48 halves = 6 chunks/row -> 384 chunks, 1 per thread
  {
    int r = tid / 6, c = tid % 6;
    const _Float16* g = qkv + (size_t)(b * HW_ + n0 + r) * O1_ + 768 + h * 48 + c * 8;
    asyncCopy16(ldsAddrOf(&lsA[r * 72 + c * 8]), g);
  }
  // B: 64 j-rows x 64 halves = 8 chunks/row -> 512 chunks
  for (int idx = tid; idx < 512; idx += 384) {
    int j = idx >> 3, c = idx & 7;
    const _Float16* g = ShT + (size_t)bh * 4096 + j * 64 + c * 8;
    asyncCopy16(ldsAddrOf(&lsB[j * 72 + c * 8]), g);
  }
  // zero pad v columns i in [48,64)
  for (int idx = tid; idx < 1024; idx += 384) {
    int n = idx >> 4, i = 48 + (idx & 15);
    lsA[n * 72 + i] = 0;
  }
  WAIT_ASYNC(0x0);
  __syncthreads();
  const _Float16* Af = (const _Float16*)lsA;
  const _Float16* Bf = (const _Float16*)lsB;
  v8f acc = {};
  #pragma unroll
  for (int kk = 0; kk < 64; kk += 32) {
    int ar = (wm * 16 + (lane & 15)) * 72 + kk;
    v16h a = cat16(*(const v8h*)(Af + ar + hi * 16),
                   *(const v8h*)(Af + ar + hi * 16 + 8));
    int br = (wj * 16 + (lane & 15)) * 72 + kk + hi * 16;
    v16h bf = cat16(*(const v8h*)(Bf + br), *(const v8h*)(Bf + br + 8));
    acc = __builtin_amdgcn_wmma_f32_16x16x32_f16(false, a, false, bf, (short)0, acc, false, false);
  }
  #pragma unroll
  for (int r = 0; r < 8; ++r) {
    size_t n = (size_t)(b * HW_ + n0 + wm * 16 + hi * 8 + r);
    attn[n * C_ + h * 48 + wj * 16 + (lane & 15)] = (_Float16)acc[r];
  }
}

// ------- output GEMM + residual: out[b,o,hw] = sum_c attn[p][c]*Wo[o][c]+x --
__global__ void __launch_bounds__(256) k_gemm_final(const _Float16* __restrict__ Wh,
                                                    const _Float16* __restrict__ Xh,
                                                    const float* __restrict__ xres,
                                                    float* __restrict__ out) {
  __shared__ __align__(16) unsigned short lsA[2][64 * 40];
  __shared__ __align__(16) unsigned short lsB[2][64 * 40];
  int tid  = threadIdx.x;
  int lane = tid & 31, wv = tid >> 5;
  int hi   = lane >> 4;
  int wm   = wv & 3, wn = wv >> 2;
  int p0   = blockIdx.x * 64;
  int o0   = blockIdx.y * 64;
  int sr = tid >> 2, sc = tid & 3;
  const _Float16* gA = Xh + (size_t)(p0 + sr) * 384 + sc * 8;
  const _Float16* gB = Wh + (size_t)(o0 + sr) * 384 + sc * 8;
  unsigned la[2] = { ldsAddrOf(&lsA[0][sr * 40 + sc * 8]),
                     ldsAddrOf(&lsA[1][sr * 40 + sc * 8]) };
  unsigned lb[2] = { ldsAddrOf(&lsB[0][sr * 40 + sc * 8]),
                     ldsAddrOf(&lsB[1][sr * 40 + sc * 8]) };
  asyncCopy16(la[0], gA);
  asyncCopy16(lb[0], gB);
  v8f acc0 = {}; v8f acc1 = {};
  for (int i = 0; i < 12; ++i) {
    if (i < 11) {
      int nb = (i + 1) & 1;
      asyncCopy16(la[nb], gA + (i + 1) * 32);
      asyncCopy16(lb[nb], gB + (i + 1) * 32);
      WAIT_ASYNC(0x2);
    } else {
      WAIT_ASYNC(0x0);
    }
    __syncthreads();
    int cb = i & 1;
    const _Float16* Af = (const _Float16*)lsA[cb];
    const _Float16* Bf = (const _Float16*)lsB[cb];
    int ar = (wm * 16 + (lane & 15)) * 40;
    v16h a = cat16(*(const v8h*)(Af + ar + hi * 16),
                   *(const v8h*)(Af + ar + hi * 16 + 8));
    int br0 = (wn * 32 + (lane & 15)) * 40 + hi * 16;
    v16h b0 = cat16(*(const v8h*)(Bf + br0), *(const v8h*)(Bf + br0 + 8));
    int br1 = (wn * 32 + 16 + (lane & 15)) * 40 + hi * 16;
    v16h b1 = cat16(*(const v8h*)(Bf + br1), *(const v8h*)(Bf + br1 + 8));
    acc0 = __builtin_amdgcn_wmma_f32_16x16x32_f16(false, a, false, b0, (short)0, acc0, false, false);
    acc1 = __builtin_amdgcn_wmma_f32_16x16x32_f16(false, a, false, b1, (short)0, acc1, false, false);
    __syncthreads();
  }
  int b = p0 >> 14;
  #pragma unroll
  for (int r = 0; r < 8; ++r) {
    int p  = p0 + wm * 16 + hi * 8 + r;
    int hw = p & (HW_ - 1);
    int o  = o0 + wn * 32 + (lane & 15);
    size_t oi = ((size_t)b * C_ + o) * HW_ + hw;
    out[oi] = acc0[r] + xres[oi];
    size_t oi2 = oi + (size_t)16 * HW_;
    out[oi2] = acc1[r] + xres[oi2];
  }
}

extern "C" void kernel_launch(void* const* d_in, const int* in_sizes, int n_in,
                              void* d_out, int out_size, void* d_ws, size_t ws_size,
                              hipStream_t stream) {
  (void)in_sizes; (void)n_in; (void)out_size; (void)ws_size;
  const float* x     = (const float*)d_in[0];
  const float* lnw   = (const float*)d_in[1];
  const float* wp    = (const float*)d_in[2];
  const float* wd    = (const float*)d_in[3];
  const float* wo    = (const float*)d_in[4];
  const float* alpha = (const float*)d_in[5];

  char* ws = (char*)d_ws;
  _Float16* xn    = (_Float16*)(ws + XN_OFF);
  _Float16* wph   = (_Float16*)(ws + WP_OFF);
  _Float16* woh   = (_Float16*)(ws + WO_OFF);
  float*    Sf    = (float*)   (ws + SF_OFF);
  _Float16* ShT   = (_Float16*)(ws + SHT_OFF);
  _Float16* qkv1  = (_Float16*)(ws + QKV1_OFF);
  _Float16* qkv2  = (_Float16*)(ws + QKV2_OFF);
  _Float16* attn  = (_Float16*)(ws + XN_OFF);   // reuse: xn dead after QKV GEMM

  k_cvt       <<<2304, 256, 0, stream>>>(wp, wo, wph, woh);
  k_ln        <<<512, 256, 0, stream>>>(x, lnw, xn);
  k_gemm_qkv  <<<dim3(2048, 18), 256, 0, stream>>>(wph, xn, qkv1);
  k_dwconv    <<<589824, 256, 0, stream>>>(qkv1, wd, qkv2);
  k_score     <<<64, 288, 0, stream>>>(qkv2, Sf);
  k_softmax   <<<16, 256, 0, stream>>>(Sf, alpha, ShT);
  k_attnv     <<<dim3(256, 64), 384, 0, stream>>>(qkv2, ShT, attn);
  k_gemm_final<<<dim3(2048, 6), 256, 0, stream>>>(woh, attn, x, (float*)d_out);
}
